// DWConvPWConv_68874095558792
// MI455X (gfx1250) — 8-bit matrix operand order repaired and checked by reading, compiled, not run on hardware
//
#include <hip/hip_runtime.h>

typedef __attribute__((ext_vector_type(8))) int v8i;

#define B_    32
#define CIN   256
#define COUT  512
#define H_    56
#define W_    56
#define HW    3136
#define NTOT  100352
#define NTILES 6272
#define MTILES 32
#define NBLK  4
#define ACT_BYTES (NTOT * CIN)

#define OFF_ACT      0
#define OFF_PWQ      (ACT_BYTES)
#define OFF_PWSCALE  (OFF_PWQ + COUT * CIN)
#define OFF_PWINV    (OFF_PWSCALE + COUT * 4)
#define OFF_PWBIAS   (OFF_PWINV + COUT * 4)
#define OFF_DWQ      (OFF_PWBIAS + COUT * 4)
#define OFF_DWINV    (OFF_DWQ + CIN * 9 * 4)
#define OFF_DWBIAS   (OFF_DWINV + CIN * 4)

__global__ void __launch_bounds__(256) pw_quant_kernel(
    const float* __restrict__ pw_w, const float* __restrict__ g,
    const float* __restrict__ be, const float* __restrict__ mu,
    const float* __restrict__ var, char* __restrict__ ws) {
  __shared__ float red[256];
  const int cout = blockIdx.x;
  const int t = threadIdx.x;
  const float w = pw_w[cout * CIN + t];
  red[t] = fabsf(w);
  __syncthreads();
  for (int s = 128; s > 0; s >>= 1) {
    if (t < s) red[t] = fmaxf(red[t], red[t + s]);
    __syncthreads();
  }
  const float scale = fmaxf(red[0] / 7.0f, 1e-30f);
  int q = (int)rintf(w / scale);
  q = q < -7 ? -7 : (q > 7 ? 7 : q);
  const int mblk = cout >> 4, row = cout & 15;
  const int kblk = t >> 6, kin = t & 63;
  const int laneHalf = (kin >> 4) & 1;
  const int chunk8 = (kin >> 5) & 1;
  const int off = mblk * 4096 + kblk * 1024 +
                  (laneHalf * 16 + row) * 32 + chunk8 * 16 + (kin & 15);
  ((signed char*)(ws + OFF_PWQ))[off] = (signed char)q;
  if (t == 0) {
    ((float*)(ws + OFF_PWSCALE))[cout] = scale;
    const float inv = g[cout] * rsqrtf(var[cout] + 1e-5f);
    ((float*)(ws + OFF_PWINV))[cout] = inv;
    ((float*)(ws + OFF_PWBIAS))[cout] = be[cout] - mu[cout] * inv;
  }
}

__global__ void __launch_bounds__(256) dw_prep_kernel(
    const float* __restrict__ dw_w, const float* __restrict__ g,
    const float* __restrict__ be, const float* __restrict__ mu,
    const float* __restrict__ var, char* __restrict__ ws) {
  const int c = threadIdx.x;
  float wv[9];
  float m = 0.f;
  #pragma unroll
  for (int i = 0; i < 9; i++) {
    wv[i] = dw_w[c * 9 + i];
    m = fmaxf(m, fabsf(wv[i]));
  }
  const float scale = fmaxf(m / 7.0f, 1e-30f);
  float* dwq = (float*)(ws + OFF_DWQ);
  #pragma unroll
  for (int i = 0; i < 9; i++) {
    int q = (int)rintf(wv[i] / scale);
    q = q < -7 ? -7 : (q > 7 ? 7 : q);
    dwq[c * 9 + i] = (float)q * scale;
  }
  const float inv = g[c] * rsqrtf(var[c] + 1e-5f);
  ((float*)(ws + OFF_DWINV))[c] = inv;
  ((float*)(ws + OFF_DWBIAS))[c] = be[c] - mu[c] * inv;
}

__global__ void __launch_bounds__(256) dw_conv_kernel(
    const float* __restrict__ x, const float* __restrict__ act_scale,
    char* __restrict__ ws) {
  const int tid = blockIdx.x * 256 + threadIdx.x;
  const int n = tid % NTOT;
  const int c4 = tid / NTOT;
  const int k0 = c4 * 4;
  const int b = n / HW, hw = n % HW;
  const int h = hw / W_, w = hw % W_;
  const float rstep = 15.0f / act_scale[0];
  const float* __restrict__ dwq = (const float*)(ws + OFF_DWQ);
  const float* __restrict__ dinv = (const float*)(ws + OFF_DWINV);
  const float* __restrict__ dbias = (const float*)(ws + OFF_DWBIAS);

  unsigned int pack = 0u;
  #pragma unroll
  for (int j = 0; j < 4; j++) {
    const int k = k0 + j;
    const float* __restrict__ xp = x + (long)(b * CIN + k) * HW;
    const float* __restrict__ wq = dwq + k * 9;
    float acc = 0.f;
    #pragma unroll
    for (int dy = -1; dy <= 1; dy++) {
      const int hh = h + dy;
      if (hh < 0 || hh >= H_) continue;
      #pragma unroll
      for (int dx = -1; dx <= 1; dx++) {
        const int ww = w + dx;
        if (ww < 0 || ww >= W_) continue;
        acc += xp[hh * W_ + ww] * wq[(dy + 1) * 3 + (dx + 1)];
      }
    }
    float y = acc * dinv[k] + dbias[k];
    y = fmaxf(y, 0.f);
    int q = (int)rintf(y * rstep);
    q = q > 15 ? 15 : q;
    pack |= ((unsigned)q) << (8 * j);
  }
  const int tile = n >> 4, col = n & 15;
  const int kblk = k0 >> 6, kin = k0 & 63;
  const int laneHalf = (kin >> 4) & 1;
  const int chunk = (kin >> 5) & 1;
  const long off = (long)tile * 4096 + kblk * 1024 +
                   (long)(laneHalf * 16 + col) * 32 + chunk * 16 + (kin & 15);
  *(unsigned int*)(ws + OFF_ACT + off) = pack;
}

__global__ void __launch_bounds__(256) pw_gemm_kernel(
    const char* __restrict__ ws, const float* __restrict__ dw_act_scale,
    const float* __restrict__ pw_act_scale, float* __restrict__ out) {
  const int wave = threadIdx.x >> 5;
  const int lane = threadIdx.x & 31;
  const int wt = blockIdx.x * 8 + wave;
  const int mtile = wt & (MTILES - 1);
  const int ngrp = wt >> 5;
  const int ntile0 = ngrp * NBLK;

  const char* __restrict__ aw = ws + OFF_PWQ + mtile * 4096 + lane * 32;
  const char* __restrict__ bw = ws + OFF_ACT + (long)ntile0 * 4096 + lane * 32;

  v8i acc[NBLK];
  #pragma unroll
  for (int t = 0; t < NBLK; t++) acc[t] = (v8i){0, 0, 0, 0, 0, 0, 0, 0};

  #pragma unroll
  for (int kb = 0; kb < 4; kb++) {
    __builtin_prefetch(bw + (kb + 1) * 1024, 0, 0);
    const v8i a = *(const v8i*)(aw + kb * 1024);
    #pragma unroll
    for (int t = 0; t < NBLK; t++) {
      const v8i bm = *(const v8i*)(bw + (long)t * 4096 + kb * 1024);
      acc[t] = __builtin_amdgcn_wmma_i32_16x16x64_iu8(
           true, a,  false, bm, acc[t],
           false,  false);
    }
  }

  const float s1 = dw_act_scale[0] * (1.0f / 15.0f);
  const float s2 = pw_act_scale[0] * (1.0f / 15.0f);
  const float rs2 = 1.0f / s2;
  const float* __restrict__ wscale = (const float*)(ws + OFF_PWSCALE);
  const float* __restrict__ pinv = (const float*)(ws + OFF_PWINV);
  const float* __restrict__ pbias = (const float*)(ws + OFF_PWBIAS);

  const int col = lane & 15;
  const int rbase = (lane >> 4) * 8;

  long obase[NBLK];
  #pragma unroll
  for (int t = 0; t < NBLK; t++) {
    const int n = (ntile0 + t) * 16 + col;
    const int b = n / HW, hw = n % HW;
    obase[t] = (long)b * COUT * HW + hw;
  }

  #pragma unroll
  for (int v = 0; v < 8; v++) {
    const int cout = mtile * 16 + rbase + v;
    const float sc = s1 * wscale[cout];
    const float inv = pinv[cout];
    const float bi = pbias[cout];
    const long crow = (long)cout * HW;
    #pragma unroll
    for (int t = 0; t < NBLK; t++) {
      float y = (float)acc[t][v] * sc;
      y = y * inv + bi;
      y = fmaxf(y, 0.f);
      float q = rintf(y * rs2);
      q = q > 15.f ? 15.f : q;
      out[obase[t] + crow] = q * s2;
    }
  }
}

extern "C" void kernel_launch(void* const* d_in, const int* in_sizes, int n_in,
                              void* d_out, int out_size, void* d_ws,
                              size_t ws_size, hipStream_t stream) {
  const float* x    = (const float*)d_in[0];
  const float* dw_w = (const float*)d_in[1];
  const float* dw_g = (const float*)d_in[2];
  const float* dw_b = (const float*)d_in[3];
  const float* dw_m = (const float*)d_in[4];
  const float* dw_v = (const float*)d_in[5];
  const float* dw_s = (const float*)d_in[6];
  const float* pw_w = (const float*)d_in[7];
  const float* pw_g = (const float*)d_in[8];
  const float* pw_b = (const float*)d_in[9];
  const float* pw_m = (const float*)d_in[10];
  const float* pw_v = (const float*)d_in[11];
  const float* pw_s = (const float*)d_in[12];
  char* ws = (char*)d_ws;
  float* out = (float*)d_out;

  pw_quant_kernel<<<COUT, 256, 0, stream>>>(pw_w, pw_g, pw_b, pw_m, pw_v, ws);
  dw_prep_kernel<<<1, 256, 0, stream>>>(dw_w, dw_g, dw_b, dw_m, dw_v, ws);
  dw_conv_kernel<<<(NTOT * 64) / 256, 256, 0, stream>>>(x, dw_s, ws);
  pw_gemm_kernel<<<(MTILES * (NTILES / NBLK)) / 8, 256, 0, stream>>>(
      ws, dw_s, pw_s, out);
}
